// WindowAttention_9320079032660
// MI455X (gfx1250) — matrix operand order repaired and checked by reading, not compiled, not run on hardware
//
#include <hip/hip_runtime.h>

// ---------------------------------------------------------------------------
// Swin-V2 window attention, fused per-window kernel for gfx1250 (MI455X).
// GEMMs: v_wmma_f32_16x16x32_bf16.  x staging: global_load_async_to_lds_b128.
// ---------------------------------------------------------------------------

typedef __attribute__((ext_vector_type(16))) __bf16 bf16x16;
typedef __attribute__((ext_vector_type(8)))  float  f32x8;

union Frag {
    bf16x16        v;
    unsigned       u32[8];
    unsigned short u16[16];
};

__device__ __forceinline__ unsigned short f2bf(float f) {
    union { float f; unsigned u; } x; x.f = f;
    unsigned r = x.u + 0x7FFFu + ((x.u >> 16) & 1u);   // round-to-nearest-even
    return (unsigned short)(r >> 16);
}
__device__ __forceinline__ float bf2f(unsigned short h) {
    union { unsigned u; float f; } x; x.u = ((unsigned)h) << 16; return x.f;
}

__device__ __forceinline__ f32x8 wmma_bf16(const Frag& a, const Frag& b, f32x8 c) {
    return __builtin_amdgcn_wmma_f32_16x16x32_bf16(false, a.v, false, b.v,
                                                   (short)0, c, false, false);
}

// Async DMA: 16 bytes per lane, global -> LDS, tracked by ASYNCcnt.
__device__ __forceinline__ void async_copy_b128(unsigned lds_off, const void* gptr) {
    asm volatile("global_load_async_to_lds_b128 %0, %1, off"
                 :: "v"(lds_off), "v"(gptr) : "memory");
}
__device__ __forceinline__ void wait_async0() {
    asm volatile("s_wait_asynccnt 0x0" ::: "memory");
}

// A fragment (16x32 bf16): lane m = m0+(lane&15); VGPR v:
// K = 16*(lane>>4) + 2*v (pair).  Source rows contiguous (u16, stride S).
__device__ __forceinline__ void load_a(Frag& A, const unsigned short* buf,
                                       int stride, int m0, int kbase, int lane) {
    const int m  = m0 + (lane & 15);
    const int hl = lane >> 4;
    const unsigned short* row = buf + m * stride + kbase;
#pragma unroll
    for (int vv = 0; vv < 8; ++vv) {
        int k = 16 * hl + 2 * vv;
        A.u32[vv] = *(const unsigned*)(row + k);
    }
}

// B fragment (32x16 bf16), element B[k][n] = W[n0+lane&15][kbase+k] where W is
// row-major [n][k] (i.e. we multiply by W^T).  K = 16*(lane>>4) + 2*vv (pair).
__device__ __forceinline__ void load_b_nk(Frag& B, const unsigned short* W,
                                          int ldw, int n0, int kbase, int lane) {
    const int n  = n0 + (lane & 15);
    const int hl = lane >> 4;
    const unsigned short* row = W + n * ldw + kbase;
#pragma unroll
    for (int vv = 0; vv < 8; ++vv) {
        int k = 16 * hl + 2 * vv;
        B.u32[vv] = *(const unsigned*)(row + k);
    }
}

// B fragment where source is truly [k][n] row-major (v matrix): column loads.
__device__ __forceinline__ void load_b_kn(Frag& B, const unsigned short* V,
                                          int ldv, int n0, int kbase, int lane) {
    const int n  = n0 + (lane & 15);
    const int hl = lane >> 4;
#pragma unroll
    for (int vv = 0; vv < 8; ++vv) {
        int k = kbase + 16 * hl + 2 * vv;
        B.u16[2 * vv]     = V[k * ldv + n];
        B.u16[2 * vv + 1] = V[(k + 1) * ldv + n];
    }
}

// ---------------------------------------------------------------------------
// Prep kernels
// ---------------------------------------------------------------------------
__global__ void prep_params(const float* __restrict__ qkv_w,
                            const float* __restrict__ proj_w,
                            const float* __restrict__ q_bias,
                            const float* __restrict__ v_bias,
                            const float* __restrict__ logit_scale,
                            unsigned short* __restrict__ wqkv,
                            unsigned short* __restrict__ wproj,
                            float* __restrict__ qkv_bias,
                            float* __restrict__ scales) {
    int i = blockIdx.x * 256 + threadIdx.x;
    if (i < 1152 * 384) wqkv[i]  = f2bf(qkv_w[i]);
    if (i < 384 * 384)  wproj[i] = f2bf(proj_w[i]);
    if (i < 1152) qkv_bias[i] = (i < 384) ? q_bias[i] : ((i < 768) ? 0.0f : v_bias[i - 768]);
    if (i < 12)   scales[i]   = __expf(fminf(logit_scale[i], 4.6051701860e+0f)); // ln(100)
}

__device__ __forceinline__ float coordval(int a) {
    float g = (float)(a - 6) * (8.0f / 6.0f);
    float s = (g > 0.f) ? 1.f : ((g < 0.f) ? -1.f : 0.f);
    return s * __log2f(fabsf(g) + 1.0f) * (1.0f / 3.0f);   // /log2(8)
}

__global__ void cpb_table(const float* __restrict__ w1, const float* __restrict__ b1,
                          const float* __restrict__ w2, float* __restrict__ tab) {
    for (int o = threadIdx.x; o < 169 * 12; o += blockDim.x) {
        int t = o / 12, h = o - 12 * t;
        float t0 = coordval(t / 13), t1 = coordval(t % 13);
        float acc = 0.0f;
        for (int u = 0; u < 512; ++u) {
            float hid = fmaxf(t0 * w1[2 * u] + t1 * w1[2 * u + 1] + b1[u], 0.0f);
            acc += w2[h * 512 + u] * hid;
        }
        tab[o] = acc;
    }
}

__global__ void cpb_rpb(const float* __restrict__ tab, float* __restrict__ rpb) {
    int idx = blockIdx.x * 256 + threadIdx.x;        // h*2401 + q*49 + kcol
    if (idx >= 12 * 49 * 49) return;
    int h = idx / 2401, rem = idx - h * 2401;
    int q = rem / 49, kc = rem - q * 49;
    int rpi = ((q / 7) - (kc / 7) + 6) * 13 + ((q % 7) - (kc % 7) + 6);
    float v = tab[rpi * 12 + h];
    rpb[idx] = 16.0f / (1.0f + __expf(-v));
}

// ---------------------------------------------------------------------------
// Fused main kernel: one block per window-batch (2048 blocks, 256 threads)
// ---------------------------------------------------------------------------
#define NP    64
#define CDIM  384
#define QST   1160   // qkv_s row stride (u16), 1152 + 8 pad
#define XST   392    // xs row stride (u16), 384 + 8 pad
#define XBYTES (49 * CDIM * 4)      // 75,264 B of x per window
#define STAGE_OFF 51200             // byte offset of staging strip in sc_pool
                                    // (fp32 x during S1, bf16 probs during S3)

__global__ __launch_bounds__(256)
void win_attn_main(const float* __restrict__ x, const float* __restrict__ mask,
                   const float* __restrict__ rpb, const float* __restrict__ scales,
                   const float* __restrict__ qkv_bias,
                   const unsigned short* __restrict__ wqkv,
                   const unsigned short* __restrict__ wproj,
                   const float* __restrict__ proj_b,
                   float* __restrict__ out) {
    __shared__ unsigned short qkv_s[NP * QST];      // 148,480 B : q|k|v (bf16)
    __shared__ float sc_pool[8 * 64 * 64];          // 131,072 B : scores / staging
    __shared__ float norms_s[8 * 128];              //   4,096 B : per-wave 1/|q|,1/|k|

    const int bw   = blockIdx.x;
    const int nw   = bw & 63;
    const int tid  = threadIdx.x;
    const int wave = tid >> 5;
    const int lane = tid & 31;
    const int hl   = lane >> 4;
    const int nn   = lane & 15;

    // ---- S1a: async DMA x (fp32) -> LDS staging strip (bypasses VGPRs)
    {
        const char* gx = (const char*)(x + (size_t)bw * 49 * CDIM);
        unsigned lds0 = (unsigned)(uintptr_t)sc_pool + STAGE_OFF;
        for (int chunk = tid; chunk < XBYTES / 16; chunk += 256)
            async_copy_b128(lds0 + chunk * 16, gx + chunk * 16);
        wait_async0();
    }
    __syncthreads();

    // ---- S1b: convert staged fp32 -> bf16 xs (rows >= 49 zero-padded)
    unsigned short* xs = (unsigned short*)sc_pool;
    {
        const float* stage = (const float*)((char*)sc_pool + STAGE_OFF);
        for (int idx = tid; idx < NP * CDIM; idx += 256) {
            int n = idx / CDIM, c = idx - n * CDIM;
            xs[n * XST + c] = f2bf((n < 49) ? stage[n * CDIM + c] : 0.0f);
        }
    }
    __syncthreads();

    // ---- S2: qkv = xs @ wqkv^T + bias  -> qkv_s (bf16).  Register-tiled: one
    // weight fragment per k-step feeds 4 M-tile accumulators.
    for (int nt = wave; nt < 72; nt += 8) {
        const int n0 = nt * 16;
        const float bj = qkv_bias[n0 + nn];
        f32x8 acc[4];
#pragma unroll
        for (int mt = 0; mt < 4; ++mt)
#pragma unroll
            for (int r = 0; r < 8; ++r) acc[mt][r] = bj;
        for (int ks = 0; ks < 12; ++ks) {
            Frag B;
            load_b_nk(B, wqkv, 384, n0, ks * 32, lane);
#pragma unroll
            for (int mt = 0; mt < 4; ++mt) {
                Frag A;
                load_a(A, xs, XST, mt * 16, ks * 32, lane);
                acc[mt] = wmma_bf16(A, B, acc[mt]);
            }
        }
#pragma unroll
        for (int mt = 0; mt < 4; ++mt)
#pragma unroll
            for (int r = 0; r < 8; ++r)
                qkv_s[(mt * 16 + 8 * hl + r) * QST + n0 + nn] = f2bf(acc[mt][r]);
    }
    __syncthreads();

    // ---- S3: attention.  Wave w owns heads w and w+8 (if < 12).
    float* sc          = sc_pool + wave * 4096;   // 64x64 fp32 score scratch
    unsigned short* pb = (unsigned short*)((char*)sc_pool + STAGE_OFF)
                       + wave * 4096;             // 64x64 bf16 probability buffer
    float* invq = norms_s + wave * 128;
    float* invk = invq + 64;

    for (int hi = 0; hi < 2; ++hi) {
        const int h = wave + hi * 8;
        if (h >= 12) break;
        const float hscale = scales[h];

        // row norms (post-hoc normalization of unnormalized q.k^T)
#pragma unroll
        for (int rr = 0; rr < 2; ++rr) {
            int m = lane + rr * 32;
            const unsigned short* qr = qkv_s + m * QST + h * 32;
            float sq = 0.f, sk = 0.f;
            for (int d = 0; d < 32; ++d) {
                float qv = bf2f(qr[d]);        sq += qv * qv;
                float kv = bf2f(qr[384 + d]);  sk += kv * kv;
            }
            invq[m] = 1.0f / fmaxf(__fsqrt_rn(sq), 1e-12f);
            invk[m] = 1.0f / fmaxf(__fsqrt_rn(sk), 1e-12f);
        }

        const float* rpb_h  = rpb  + h  * 49 * 49;
        const float* mask_w = mask + nw * 49 * 49;

        // s = q @ k^T: cache all 4 k^T fragments, hoist q fragment per M-tile
        Frag Bk[4];
#pragma unroll
        for (int ntn = 0; ntn < 4; ++ntn)
            load_b_nk(Bk[ntn], qkv_s, QST, ntn * 16, 384 + h * 32, lane);
        for (int mt = 0; mt < 4; ++mt) {
            Frag A;
            load_a(A, qkv_s, QST, mt * 16, h * 32, lane);
#pragma unroll
            for (int ntn = 0; ntn < 4; ++ntn) {
                f32x8 c;
#pragma unroll
                for (int r = 0; r < 8; ++r) c[r] = 0.0f;
                c = wmma_bf16(A, Bk[ntn], c);
#pragma unroll
                for (int r = 0; r < 8; ++r) {
                    int m = mt * 16 + 8 * hl + r;
                    int n = ntn * 16 + nn;
                    float val = -1e30f;
                    if (m < 49 && n < 49)
                        val = c[r] * invq[m] * invk[n] * hscale
                            + rpb_h[m * 49 + n] + mask_w[m * 49 + n];
                    sc[m * 64 + n] = val;
                }
            }
        }

        // softmax per row; probabilities written ONCE as bf16 to pb
        // (cols >= 49 hold -1e30 -> exp == 0)
        for (int m = 0; m < 64; ++m) {
            float v0 = sc[m * 64 + lane];
            float v1 = sc[m * 64 + 32 + lane];
            float mx = fmaxf(v0, v1);
#pragma unroll
            for (int off = 16; off; off >>= 1) mx = fmaxf(mx, __shfl_xor(mx, off, 32));
            float e0 = __expf(v0 - mx), e1 = __expf(v1 - mx);
            float s = e0 + e1;
#pragma unroll
            for (int off = 16; off; off >>= 1) s += __shfl_xor(s, off, 32);
            float inv = 1.0f / fmaxf(s, 1e-30f);
            pb[m * 64 + lane]      = f2bf(e0 * inv);
            pb[m * 64 + 32 + lane] = f2bf(e1 * inv);
        }

        // out_h = p @ v: hoist the scalar-gather v fragments, 8 live accumulators
        f32x8 acc[4][2];
#pragma unroll
        for (int mt = 0; mt < 4; ++mt)
#pragma unroll
            for (int dt = 0; dt < 2; ++dt)
#pragma unroll
                for (int r = 0; r < 8; ++r) acc[mt][dt][r] = 0.0f;
        for (int kt = 0; kt < 2; ++kt) {
            Frag Bv[2];
            load_b_kn(Bv[0], qkv_s + 768 + h * 32, QST, 0,  kt * 32, lane);
            load_b_kn(Bv[1], qkv_s + 768 + h * 32, QST, 16, kt * 32, lane);
#pragma unroll
            for (int mt = 0; mt < 4; ++mt) {
                Frag A;
                load_a(A, pb, 64, mt * 16, kt * 32, lane);   // bf16 probs, b128 loads
                acc[mt][0] = wmma_bf16(A, Bv[0], acc[mt][0]);
                acc[mt][1] = wmma_bf16(A, Bv[1], acc[mt][1]);
            }
        }
        // write bf16 head output into the (dead) q columns of this head
#pragma unroll
        for (int mt = 0; mt < 4; ++mt)
#pragma unroll
            for (int dt = 0; dt < 2; ++dt)
#pragma unroll
                for (int r = 0; r < 8; ++r)
                    qkv_s[(mt * 16 + 8 * hl + r) * QST + h * 32 + dt * 16 + nn]
                        = f2bf(acc[mt][dt][r]);
    }
    __syncthreads();

    // ---- S4: proj.  Activation = qkv_s cols [0,384) (attention output, bf16)
    for (int nt = wave; nt < 24; nt += 8) {
        const int n0 = nt * 16;
        const int j  = n0 + nn;
        const float bj = proj_b[j];
        f32x8 acc[4];
#pragma unroll
        for (int mt = 0; mt < 4; ++mt)
#pragma unroll
            for (int r = 0; r < 8; ++r) acc[mt][r] = bj;
        for (int ks = 0; ks < 12; ++ks) {
            Frag B;
            load_b_nk(B, wproj, 384, n0, ks * 32, lane);
#pragma unroll
            for (int mt = 0; mt < 4; ++mt) {
                Frag A;
                load_a(A, qkv_s, QST, mt * 16, ks * 32, lane);
                acc[mt] = wmma_bf16(A, B, acc[mt]);
            }
        }
#pragma unroll
        for (int mt = 0; mt < 4; ++mt)
#pragma unroll
            for (int r = 0; r < 8; ++r) {
                int m = mt * 16 + 8 * hl + r;
                if (m < 49)
                    out[((size_t)bw * 49 + m) * 384 + j] = acc[mt][r];
            }
    }
}

// ---------------------------------------------------------------------------
// Workspace layout (bytes)
// ---------------------------------------------------------------------------
static const size_t OFF_SCALE = 0;        // 12 f32
static const size_t OFF_BIAS  = 256;      // 1152 f32
static const size_t OFF_TAB   = 4864;     // 169*12 f32
static const size_t OFF_RPB   = 13056;    // 12*49*49 f32
static const size_t OFF_WQKV  = 128384;   // 1152*384 bf16
static const size_t OFF_WPROJ = 1013120;  // 384*384 bf16   (total ~1.31 MB)

extern "C" void kernel_launch(void* const* d_in, const int* in_sizes, int n_in,
                              void* d_out, int out_size, void* d_ws, size_t ws_size,
                              hipStream_t stream) {
    const float* x           = (const float*)d_in[0];
    const float* mask        = (const float*)d_in[1];
    const float* qkv_w       = (const float*)d_in[2];
    const float* q_bias      = (const float*)d_in[3];
    const float* v_bias      = (const float*)d_in[4];
    const float* logit_scale = (const float*)d_in[5];
    const float* cpb_w1      = (const float*)d_in[6];
    const float* cpb_b1      = (const float*)d_in[7];
    const float* cpb_w2      = (const float*)d_in[8];
    const float* proj_w      = (const float*)d_in[9];
    const float* proj_b      = (const float*)d_in[10];
    float* out = (float*)d_out;
    char*  ws  = (char*)d_ws;

    float*          scales   = (float*)(ws + OFF_SCALE);
    float*          qkv_bias = (float*)(ws + OFF_BIAS);
    float*          tab      = (float*)(ws + OFF_TAB);
    float*          rpb      = (float*)(ws + OFF_RPB);
    unsigned short* wqkv     = (unsigned short*)(ws + OFF_WQKV);
    unsigned short* wproj    = (unsigned short*)(ws + OFF_WPROJ);

    prep_params<<<1728, 256, 0, stream>>>(qkv_w, proj_w, q_bias, v_bias,
                                          logit_scale, wqkv, wproj, qkv_bias, scales);
    cpb_table<<<1, 256, 0, stream>>>(cpb_w1, cpb_b1, cpb_w2, tab);
    cpb_rpb<<<113, 256, 0, stream>>>(tab, rpb);
    win_attn_main<<<2048, 256, 0, stream>>>(x, mask, rpb, scales, qkv_bias,
                                            wqkv, wproj, proj_b, out);
}
